// ScaledDotProductAttention_35596688949467
// MI455X (gfx1250) — matrix operand order repaired and checked by reading, not compiled, not run on hardware
//
#include <hip/hip_runtime.h>

typedef __attribute__((ext_vector_type(16))) __bf16 v16bf;
typedef __attribute__((ext_vector_type(8)))  float  v8f;
typedef unsigned short bfraw;

__device__ __forceinline__ bfraw f2bf(float f) {
    union { float f; unsigned u; } c; c.f = f;
    unsigned r = c.u + 0x7FFFu + ((c.u >> 16) & 1u);   // round-to-nearest-even
    return (bfraw)(r >> 16);
}

// ---------------------------------------------------------------- convert f32 -> bf16
__global__ void cvt_bf16_kernel(const float* __restrict__ src, bfraw* __restrict__ dst, size_t n) {
    size_t i = (size_t)blockIdx.x * blockDim.x + threadIdx.x;
    size_t stride = (size_t)gridDim.x * blockDim.x;
    for (; i < n; i += stride) dst[i] = f2bf(src[i]);
}

// ---------------------------------------------------------------- WMMA GEMM: C = scale*(A @ B^T) + bias
// A: [M,K] bf16 row-major, B: [N,K] bf16 row-major.
// mode 0: C f32 [M,N]; mode 1: C bf16 [M,N]; mode 2: C bf16 transposed [N,M]
#define BM 128
#define BN 128
#define BK 32
#define LDA_S 40   // padded LDS row stride (20-bank stride -> conflict-free; 80B row = 16B aligned)
#define LDB_S 40

union FragBF { v16bf v; uint4 u[2]; };

// issue one 32-byte row-chunk: LDS[lds+0..31] = MEM[gp+0..31] (async, ASYNCcnt)
__device__ __forceinline__ void async_copy32(unsigned lds, const bfraw* gp) {
    asm volatile(
        "global_load_async_to_lds_b128 %0, %1, off\n\t"
        "global_load_async_to_lds_b128 %0, %1, off offset:16"
        :: "v"(lds), "v"(gp) : "memory");
}

__global__ __launch_bounds__(256) void gemm_bf16_wmma_kernel(
    const bfraw* __restrict__ A, const bfraw* __restrict__ B,
    const float* __restrict__ bias,
    float* __restrict__ Cf, bfraw* __restrict__ Cb,
    int M, int N, int K, float scale, int mode)
{
    __shared__ bfraw As[2][BM * LDA_S];
    __shared__ bfraw Bs[2][BN * LDB_S];

    const int tid  = threadIdx.x;
    const int lane = tid & 31;
    const int wave = tid >> 5;
    const int mw   = wave & 3;   // 4 waves along M
    const int nw   = wave >> 2;  // 2 waves along N
    const int bMo  = blockIdx.y * BM;
    const int bNo  = blockIdx.x * BN;

    // cooperative tile load mapping: 256 threads x 16 bf16 = 128x32 tile
    const int lrow = tid >> 1;        // 0..127
    const int lcol = (tid & 1) * 16;  // 0 or 16

    const bfraw* ga = A + (size_t)(bMo + lrow) * K + lcol;  // this thread's A source row
    const bfraw* gb = B + (size_t)(bNo + lrow) * K + lcol;  // this thread's B source row

    // LDS byte offsets for this thread's async destinations (low 32 bits of generic ptr)
    unsigned ldsA[2], ldsB[2];
    #pragma unroll
    for (int b = 0; b < 2; ++b) {
        ldsA[b] = (unsigned)(unsigned long long)(void*)&As[b][lrow * LDA_S + lcol];
        ldsB[b] = (unsigned)(unsigned long long)(void*)&Bs[b][lrow * LDB_S + lcol];
    }

    v8f acc[2][4] = {};  // zero-init accumulators

    // fragment index precompute
    const int arow  = lane & 15;
    const int aoff  = (lane >> 4) * 8;
    const int bncol = lane & 15;
    const int bkoff = (lane >> 4) * 8;

    // one K-tile of compute from LDS buffer `cur`
    auto compute_tile = [&](int cur) {
        // A fragments (ISA 7.12.2 bf16 A 16x32): lanes 0-15 -> K 0-7 & 16-23; lanes 16-31 -> K 8-15 & 24-31
        FragBF afr[2];
        #pragma unroll
        for (int i = 0; i < 2; ++i) {
            const int r = mw * 32 + i * 16 + arow;
            afr[i].u[0] = *(const uint4*)&As[cur][r * LDA_S + aoff];
            afr[i].u[1] = *(const uint4*)&As[cur][r * LDA_S + aoff + 16];
        }
        // B fragments: lane = column (lanes 0-15: K 0-7 & 16-23, lanes 16-31: K 8-15 & 24-31)
        FragBF bfr[4];
        #pragma unroll
        for (int j = 0; j < 4; ++j) {
            const int n = nw * 64 + j * 16 + bncol;
            bfr[j].u[0] = *(const uint4*)&Bs[cur][n * LDB_S + bkoff];
            bfr[j].u[1] = *(const uint4*)&Bs[cur][n * LDB_S + bkoff + 16];
        }
        #pragma unroll
        for (int i = 0; i < 2; ++i)
            #pragma unroll
            for (int j = 0; j < 4; ++j)
                acc[i][j] = __builtin_amdgcn_wmma_f32_16x16x32_bf16(
                    false, afr[i].v, false, bfr[j].v, (short)0, acc[i][j], false, false);
    };

    const int nk = K / BK;  // >= 2 for all invocations here

    // prologue: fetch tile 0 into buffer 0
    async_copy32(ldsA[0], ga);
    async_copy32(ldsB[0], gb);

    const bfraw* gan = ga + BK;
    const bfraw* gbn = gb + BK;

    // steady state: unconditional prefetch of tile kt+1, compute tile kt
    for (int kt = 0; kt < nk - 1; ++kt) {
        const int cur = kt & 1;
        async_copy32(ldsA[1 - cur], gan);
        async_copy32(ldsB[1 - cur], gbn);
        gan += BK;
        gbn += BK;
        // ASYNCcnt in-order: <=4 outstanding means current buffer's 4 are complete
        asm volatile("s_wait_asynccnt 0x4" ::: "memory");
        __syncthreads();                       // current buffer visible to all waves
        compute_tile(cur);
        __syncthreads();                       // all reads of `cur` done before it is async-overwritten
    }

    // final tile (no prefetch, no trailing barrier needed)
    asm volatile("s_wait_asynccnt 0x0" ::: "memory");
    __syncthreads();
    compute_tile((nk - 1) & 1);

    // epilogue: C fragment layout (lanes 0-15: M=vgpr, lanes 16-31: M=vgpr+8; N=lane&15)
    #pragma unroll
    for (int i = 0; i < 2; ++i) {
        #pragma unroll
        for (int j = 0; j < 4; ++j) {
            const int rowBase = bMo + mw * 32 + i * 16 + (lane >> 4) * 8;
            const int col     = bNo + nw * 64 + j * 16 + (lane & 15);
            const float bc = bias ? bias[col] : 0.0f;
            #pragma unroll
            for (int e = 0; e < 8; ++e) {
                const int r = rowBase + e;
                const float val = acc[i][j][e] * scale + bc;
                if (mode == 0)      Cf[(size_t)r * N + col] = val;
                else if (mode == 1) Cb[(size_t)r * N + col] = f2bf(val);
                else                Cb[(size_t)col * M + r] = f2bf(val);
            }
        }
    }
}

// ---------------------------------------------------------------- row softmax: f32 [rows,4096] -> bf16 P
__global__ __launch_bounds__(256) void softmax_rows_kernel(
    const float* __restrict__ S, bfraw* __restrict__ P, int cols /*==4096*/)
{
    const int row = blockIdx.x;
    const int t   = threadIdx.x;
    const float* src = S + (size_t)row * cols;

    float v[16];
    float lm = -3.402823466e38f;
    #pragma unroll
    for (int i = 0; i < 16; ++i) { v[i] = src[i * 256 + t]; lm = fmaxf(lm, v[i]); }

    __shared__ float red[256];
    red[t] = lm; __syncthreads();
    for (int s = 128; s > 0; s >>= 1) { if (t < s) red[t] = fmaxf(red[t], red[t + s]); __syncthreads(); }
    const float m = red[0]; __syncthreads();

    float ls = 0.0f;
    #pragma unroll
    for (int i = 0; i < 16; ++i) { v[i] = __expf(v[i] - m); ls += v[i]; }
    red[t] = ls; __syncthreads();
    for (int s = 128; s > 0; s >>= 1) { if (t < s) red[t] += red[t + s]; __syncthreads(); }
    const float inv = 1.0f / red[0];

    bfraw* dst = P + (size_t)row * cols;
    #pragma unroll
    for (int i = 0; i < 16; ++i) dst[i * 256 + t] = f2bf(v[i] * inv);
}

// ---------------------------------------------------------------- launcher
extern "C" void kernel_launch(void* const* d_in, const int* in_sizes, int n_in,
                              void* d_out, int out_size, void* d_ws, size_t ws_size,
                              hipStream_t stream)
{
    (void)in_sizes; (void)n_in; (void)out_size; (void)ws_size;
    const int Nx = 4096, Ny = 4096, K = 1024, H = 1024;  // x_in = y_in = hidden = out_dim = 1024

    const float* x  = (const float*)d_in[0];
    const float* y  = (const float*)d_in[1];
    const float* Wq = (const float*)d_in[2];
    const float* bq = (const float*)d_in[3];
    const float* Wk = (const float*)d_in[4];
    const float* bk = (const float*)d_in[5];
    const float* Wv = (const float*)d_in[6];
    const float* bv = (const float*)d_in[7];
    float* out = (float*)d_out;

    char* ws = (char*)d_ws;
    const size_t MB = 1024ull * 1024ull;
    bfraw* x16  = (bfraw*)(ws + 0 * MB);    //  8 MB  [4096,1024]
    bfraw* y16  = (bfraw*)(ws + 8 * MB);    //  8 MB  [4096,1024]
    bfraw* wq16 = (bfraw*)(ws + 16 * MB);   //  2 MB  [1024,1024]
    bfraw* wk16 = (bfraw*)(ws + 18 * MB);   //  2 MB
    bfraw* wv16 = (bfraw*)(ws + 20 * MB);   //  2 MB
    bfraw* q16  = (bfraw*)(ws + 22 * MB);   //  8 MB  [4096,1024]
    bfraw* k16  = (bfraw*)(ws + 30 * MB);   //  8 MB  [4096,1024]
    bfraw* vT16 = (bfraw*)(ws + 38 * MB);   //  8 MB  [1024,4096] (v transposed)
    float* scor = (float*)(ws + 46 * MB);   // 64 MB  [4096,4096] (fits in 192 MB L2)
    bfraw* P16  = (bfraw*)(ws + 0 * MB);    // 32 MB, aliases x16/y16/w*/q16/k16 (all dead by then)

    // 1) f32 -> bf16 conversions
    cvt_bf16_kernel<<<2048, 256, 0, stream>>>(x,  x16,  (size_t)Nx * K);
    cvt_bf16_kernel<<<2048, 256, 0, stream>>>(y,  y16,  (size_t)Ny * K);
    cvt_bf16_kernel<<<1024, 256, 0, stream>>>(Wq, wq16, (size_t)H * K);
    cvt_bf16_kernel<<<1024, 256, 0, stream>>>(Wk, wk16, (size_t)H * K);
    cvt_bf16_kernel<<<1024, 256, 0, stream>>>(Wv, wv16, (size_t)H * K);

    // 2) projections: q = x@Wq^T + bq (bf16), k = y@Wk^T + bk (bf16), vT = (y@Wv^T + bv)^T (bf16)
    dim3 gProj(H / BN, Nx / BM);
    gemm_bf16_wmma_kernel<<<gProj, 256, 0, stream>>>(x16, wq16, bq, nullptr, q16,  Nx, H, K, 1.0f, 1);
    gemm_bf16_wmma_kernel<<<gProj, 256, 0, stream>>>(y16, wk16, bk, nullptr, k16,  Ny, H, K, 1.0f, 1);
    gemm_bf16_wmma_kernel<<<gProj, 256, 0, stream>>>(y16, wv16, bv, nullptr, vT16, Ny, H, K, 1.0f, 2);

    // 3) scores = (q @ k^T) / sqrt(H)  (f32)
    dim3 gScore(Ny / BN, Nx / BM);
    gemm_bf16_wmma_kernel<<<gScore, 256, 0, stream>>>(q16, k16, nullptr, scor, nullptr,
                                                      Nx, Ny, H, 0.03125f, 0);

    // 4) row softmax -> P bf16
    softmax_rows_kernel<<<Nx, 256, 0, stream>>>(scor, P16, Ny);

    // 5) out = P @ v = P @ (vT)^T  (f32 to d_out)
    dim3 gOut(H / BN, Nx / BM);
    gemm_bf16_wmma_kernel<<<gOut, 256, 0, stream>>>(P16, vT16, nullptr, out, nullptr,
                                                    Nx, H, Ny, 1.0f, 0);
}
